// GATv2_65910568125059
// MI455X (gfx1250) — matrix operand order repaired and checked by reading, not compiled, not run on hardware
//
#include <hip/hip_runtime.h>
#include <hip/hip_bf16.h>

// ---------------------------------------------------------------------------
// GATv2 attention for MI455X (gfx1250, wave32, WMMA)
//   B=2, T=1024, C_IN=128, D=64
//   k1: q/k projections (f32 VALU) + feat -> f16 transposed copy
//   k2: relu-scores + mask + softmax (f32 VALU, wave-per-row) -> att (f16)
//   k3: att @ feat via v_wmma_f32_16x16x32_f16 (f32 accumulate)
// ---------------------------------------------------------------------------

#define Bn   2
#define Tn   1024
#define CIN  128
#define Dn   64
#define NEG_INF_F (-1.0e22f)

typedef _Float16 v16h __attribute__((ext_vector_type(16)));
typedef _Float16 v8h  __attribute__((ext_vector_type(8)));
typedef float    v8f  __attribute__((ext_vector_type(8)));

// ---------------------------------------------------------------------------
// Kernel 1: projections + f16 transposed feat
//   grid = B*T blocks, 128 threads.
//   thread t<64  -> q[row][t]   = dot(feat_row, W1[t])
//   thread t>=64 -> k[row][t-64]= dot(feat_row, W2[t-64])
//   also featT_f16[b][c][t] = (f16)feat[b][t][c]
// ---------------------------------------------------------------------------
__global__ void gat_proj_kernel(const float* __restrict__ feat,
                                const float* __restrict__ W1,
                                const float* __restrict__ W2,
                                float* __restrict__ q,
                                float* __restrict__ k,
                                _Float16* __restrict__ featT) {
    const int row = blockIdx.x;          // b*T + t
    const int b   = row >> 10;
    const int t   = row & (Tn - 1);
    const int tid = threadIdx.x;         // 0..127

    __shared__ __align__(16) float frow[CIN];

    const float v = feat[(size_t)row * CIN + tid];
    frow[tid] = v;
    featT[((size_t)(b * CIN + tid)) * Tn + t] = (_Float16)v;
    __syncthreads();

    const float* Wr = (tid < Dn) ? (W1 + (size_t)tid * CIN)
                                 : (W2 + (size_t)(tid - Dn) * CIN);
    const float4* W4 = (const float4*)Wr;
    const float4* f4 = (const float4*)frow;

    float s = 0.0f;
#pragma unroll
    for (int d4 = 0; d4 < CIN / 4; ++d4) {
        const float4 wv = W4[d4];
        const float4 fv = f4[d4];
        s = fmaf(wv.x, fv.x, s);
        s = fmaf(wv.y, fv.y, s);
        s = fmaf(wv.z, fv.z, s);
        s = fmaf(wv.w, fv.w, s);
    }
    if (tid < Dn) q[(size_t)row * Dn + tid]        = s;
    else          k[(size_t)row * Dn + (tid - Dn)] = s;
}

// ---------------------------------------------------------------------------
// Kernel 2: scores + mask + softmax -> att (f16)
//   grid = (T/8, B), 256 threads (8 waves). Wave w owns row i = i0 + w.
//   q tiles (64 rows x 64 d, padded to 65) staged in LDS cooperatively.
// ---------------------------------------------------------------------------
#define TI 8
#define TJ 64

__global__ void gat_att_kernel(const float* __restrict__ q,
                               const float* __restrict__ kk,
                               const float* __restrict__ avec,
                               const float* __restrict__ adj,
                               _Float16* __restrict__ att) {
    const int b   = blockIdx.y;
    const int i0  = blockIdx.x * TI;
    const int tid = threadIdx.x;         // 0..255
    const int w   = tid >> 5;            // wave id == local row
    const int l   = tid & 31;            // lane

    __shared__ float q_lds[TJ][Dn + 1];  // +1 pad: kill stride-64 bank conflicts
    __shared__ float k_lds[TI][Dn];
    __shared__ float a_lds[Dn];
    __shared__ float sc[TI][Tn];         // 32 KB score rows

    // stage k rows for this i-tile and the attention vector a
    for (int idx = tid; idx < TI * Dn; idx += 256)
        k_lds[idx >> 6][idx & 63] = kk[((size_t)(b * Tn + i0)) * Dn + idx];
    if (tid < Dn) a_lds[tid] = avec[tid];

    const int i = i0 + w;
    const float* adjrow = adj + ((size_t)(b * Tn + i)) * Tn;

    for (int j0 = 0; j0 < Tn; j0 += TJ) {
        __syncthreads();
        for (int idx = tid; idx < TJ * Dn; idx += 256) {
            const int jj = idx >> 6, d = idx & 63;
            q_lds[jj][d] = q[((size_t)(b * Tn + j0 + jj)) * Dn + d];
        }
        __syncthreads();

        for (int jj = l; jj < TJ; jj += 32) {
            float s = 0.0f;
#pragma unroll
            for (int d = 0; d < Dn; ++d) {
                float e = q_lds[jj][d] + k_lds[w][d];
                e = fmaxf(e, 0.0f);                 // ReLU
                s = fmaf(e, a_lds[d], s);
            }
            const int j = j0 + jj;
            const float msk = (adjrow[j] > 0.0f) ? 0.0f : NEG_INF_F;
            sc[w][j] = s + msk;
        }
    }

    // wave-local softmax over sc[w][0..1023] (wave32 shuffles)
    float m = -3.0e38f;
    for (int j = l; j < Tn; j += 32) m = fmaxf(m, sc[w][j]);
#pragma unroll
    for (int off = 16; off > 0; off >>= 1) m = fmaxf(m, __shfl_xor(m, off, 32));

    float sum = 0.0f;
    for (int j = l; j < Tn; j += 32) {
        const float e = expf(sc[w][j] - m);
        sc[w][j] = e;
        sum += e;
    }
#pragma unroll
    for (int off = 16; off > 0; off >>= 1) sum += __shfl_xor(sum, off, 32);
    const float inv = 1.0f / sum;

    _Float16* arow = att + ((size_t)(b * Tn + i)) * Tn;
    for (int j = l; j < Tn; j += 32) arow[j] = (_Float16)(sc[w][j] * inv);
}

// ---------------------------------------------------------------------------
// Kernel 3: out[b] = att[b] (1024x1024, f16) @ feat[b] (1024x128, f16), f32 acc
//   grid = (T/16, B), 256 threads (8 waves). Wave w -> 16x16 tile at
//   (i0, c0 = w*16). K loop in steps of 32 with v_wmma_f32_16x16x32_f16.
//   Fragment layouts used here (A/B in contiguous k order, not the ISA table order):
//     A: lane M = l&15, elem e -> K = hi*16 + e   (contiguous)
//     B: lane N = l&15, elem e -> K = hi*16 + e   (contiguous; featT is [c][j])
//     C/D: lane N = l&15, vgpr r -> M = r + 8*hi
// ---------------------------------------------------------------------------
__global__ void gat_out_wmma_kernel(const _Float16* __restrict__ att,
                                    const _Float16* __restrict__ featT,
                                    float* __restrict__ out) {
    const int b  = blockIdx.y;
    const int i0 = blockIdx.x << 4;
    const int w  = threadIdx.x >> 5;     // column tile
    const int l  = threadIdx.x & 31;
    const int m  = l & 15;
    const int hi = l >> 4;
    const int c0 = w << 4;

    const _Float16* Arow = att   + ((size_t)(b * Tn  + i0 + m)) * Tn + hi * 16;
    const _Float16* Bcol = featT + ((size_t)(b * CIN + c0 + m)) * Tn + hi * 16;

    v8f acc = {};
#pragma unroll 4
    for (int k0 = 0; k0 < Tn; k0 += 32) {
        const v8h alo = *(const v8h*)(Arow + k0);        // K = hi*16 + 0..7
        const v8h ahi = *(const v8h*)(Arow + k0 + 8);    // K = hi*16 + 8..15
        v16h av;
#pragma unroll
        for (int e = 0; e < 8; ++e) { av[e] = alo[e]; av[e + 8] = ahi[e]; }

        const v8h blo = *(const v8h*)(Bcol + k0);        // K = hi*16 + 0..7
        const v8h bhi = *(const v8h*)(Bcol + k0 + 8);    // K = hi*16 + 8..15
        v16h bv;
#pragma unroll
        for (int e = 0; e < 8; ++e) { bv[e] = blo[e]; bv[e + 8] = bhi[e]; }

        acc = __builtin_amdgcn_wmma_f32_16x16x32_f16(
            /*neg_a=*/false, av, /*neg_b=*/false, bv,
            /*c_mod=*/(short)0, acc, /*reuse_a=*/false, /*reuse_b=*/false);
    }

    float* orow = out + ((size_t)(b * Tn + i0 + hi * 8)) * CIN + c0 + m;
#pragma unroll
    for (int r = 0; r < 8; ++r) orow[(size_t)r * CIN] = acc[r];
}

// ---------------------------------------------------------------------------
extern "C" void kernel_launch(void* const* d_in, const int* in_sizes, int n_in,
                              void* d_out, int out_size, void* d_ws, size_t ws_size,
                              hipStream_t stream) {
    const float* feat = (const float*)d_in[0];   // [B,T,CIN]
    const float* adj  = (const float*)d_in[1];   // [B,T,T]
    const float* W1   = (const float*)d_in[2];   // [D,CIN]
    const float* W2   = (const float*)d_in[3];   // [D,CIN]
    const float* a    = (const float*)d_in[4];   // [D]
    float* out = (float*)d_out;                  // [B,T,CIN]

    char* ws = (char*)d_ws;
    float*     q     = (float*)(ws);                        // B*T*D f32   = 512 KB
    float*     k     = (float*)(ws + (512u << 10));         // B*T*D f32   = 512 KB
    _Float16*  featT = (_Float16*)(ws + (1024u << 10));     // B*CIN*T f16 = 512 KB
    _Float16*  att   = (_Float16*)(ws + (1536u << 10));     // B*T*T  f16  = 4 MB

    gat_proj_kernel<<<dim3(Bn * Tn), dim3(CIN), 0, stream>>>(feat, W1, W2, q, k, featT);
    gat_att_kernel<<<dim3(Tn / TI, Bn), dim3(256), 0, stream>>>(q, k, a, adj, att);
    gat_out_wmma_kernel<<<dim3(Tn / 16, Bn), dim3(256), 0, stream>>>(att, featT, out);
}
